// SimpleTacticsGNN_7327214207186
// MI455X (gfx1250) — matrix operand order repaired and checked by reading, not compiled, not run on hardware
//
#include <hip/hip_runtime.h>
#include <math.h>

// ---------------------------------------------------------------------------
// Problem constants (match reference)
// ---------------------------------------------------------------------------
constexpr int N_NODES = 50000;
constexpr int N_EDGES = 800000;
constexpr int FIN     = 128;
constexpr int H       = 4;
constexpr int C       = 32;
constexpr int HID     = H * C;   // 128
constexpr int FE      = 16;
constexpr int NB      = 8;       // graphs per batch
constexpr float NEG_SLOPE = 0.2f;

typedef __attribute__((ext_vector_type(16))) _Float16 v16h;
typedef __attribute__((ext_vector_type(8)))  float    v8f;

// ---------------------------------------------------------------------------
// float atomic max via integer punning (works across sign mix)
// ---------------------------------------------------------------------------
__device__ inline void atomicMaxFloat(float* addr, float val) {
    if (val >= 0.0f) {
        atomicMax((int*)addr, __float_as_int(val));
    } else {
        atomicMin((unsigned int*)addr, (unsigned int)__float_as_int(val));
    }
}

// ---------------------------------------------------------------------------
// Pre-swizzle a 128x128 f32 weight matrix into packed f16 WMMA B-fragments.
// Layout: Wp[(kk*8 + t)*32 + lane][0..15] halfs, where
//   kk = k-step (K=32 chunk), t = 16-col tile, lane = wave lane.
// B (32x16 f16) fragment: lane l -> col (l&15); low half-wave K {0..7,16..23},
// high half-wave K {8..15,24..31} (matches A fragment order).
// One thread per (kk,t,lane) writes 16 contiguous halfs (32B).
// ---------------------------------------------------------------------------
__global__ void weight_swizzle(const float* __restrict__ W, _Float16* __restrict__ Wp)
{
    int idx = blockIdx.x * blockDim.x + threadIdx.x;   // 4*8*32 = 1024
    if (idx >= 4 * 8 * 32) return;
    int lane = idx & 31;
    int t    = (idx >> 5) & 7;
    int kk   = idx >> 8;
    int col16  = lane & 15;
    int hiHalf = lane >> 4;
    int n  = t * 16 + col16;
    int kb = kk * 32 + hiHalf * 8;
    _Float16* dst = Wp + (size_t)idx * 16;
    #pragma unroll
    for (int i = 0; i < 16; ++i)
        dst[i] = (_Float16)W[(size_t)(kb + i + (i >= 8 ? 8 : 0)) * HID + n];
}

// ---------------------------------------------------------------------------
// Dual-output GEMM: Yl = X@Wl + bl, Yr = X@Wr + br  (M x 128, K = 128)
// via v_wmma_f32_16x16x32_f16. One wave computes a 16-row x 128-col panel
// for BOTH outputs, sharing A fragments (64 WMMAs / wave).
// A (16x32 f16): lane l -> row (l&15); low half-wave holds K {0..7,16..23},
//                high half-wave holds K {8..15,24..31}.
// C/D (16x16 f32): lane l -> col (l&15); vgpr v -> row v + 8*(l>=16).
// ---------------------------------------------------------------------------
__global__ void __launch_bounds__(128)
gemm_node_wmma_dual(const float* __restrict__ X,
                    const _Float16* __restrict__ Wpl, const _Float16* __restrict__ Wpr,
                    const float* __restrict__ bl, const float* __restrict__ br,
                    float* __restrict__ Yl, float* __restrict__ Yr, int M)
{
    const int K = 128;
    int lane = threadIdx.x & 31;
    int wave = threadIdx.x >> 5;
    int tm   = blockIdx.x * 4 + wave;          // 16-row tile index (uniform per wave)
    if (tm * 16 >= M) return;                   // M is a multiple of 16 here

    int col16  = lane & 15;
    int hiHalf = lane >> 4;                     // 0 or 1
    int row    = tm * 16 + col16;               // A row for this lane

    v8f accl[8] = {};
    v8f accr[8] = {};

    const float* xrow = X + (size_t)row * K;
    const v16h* fl = (const v16h*)Wpl;          // 32B-aligned fragments
    const v16h* fr = (const v16h*)Wpr;

    #pragma unroll
    for (int kk = 0; kk < 4; ++kk) {
        int k0 = kk * 32;
        int base8 = hiHalf * 8;
        // ---- A fragment: 4 aligned float4 loads + pack to f16 ----
        float4 a0 = *(const float4*)(xrow + k0 + base8);
        float4 a1 = *(const float4*)(xrow + k0 + base8 + 4);
        float4 a2 = *(const float4*)(xrow + k0 + 16 + base8);
        float4 a3 = *(const float4*)(xrow + k0 + 16 + base8 + 4);
        v16h af;
        af[0]  = (_Float16)a0.x; af[1]  = (_Float16)a0.y;
        af[2]  = (_Float16)a0.z; af[3]  = (_Float16)a0.w;
        af[4]  = (_Float16)a1.x; af[5]  = (_Float16)a1.y;
        af[6]  = (_Float16)a1.z; af[7]  = (_Float16)a1.w;
        af[8]  = (_Float16)a2.x; af[9]  = (_Float16)a2.y;
        af[10] = (_Float16)a2.z; af[11] = (_Float16)a2.w;
        af[12] = (_Float16)a3.x; af[13] = (_Float16)a3.y;
        af[14] = (_Float16)a3.z; af[15] = (_Float16)a3.w;

        // ---- 8 column tiles for each output ----
        #pragma unroll
        for (int t = 0; t < 8; ++t) {
            v16h bfl = fl[(kk * 8 + t) * 32 + lane];   // contiguous 32B/lane
            accl[t] = __builtin_amdgcn_wmma_f32_16x16x32_f16(
                          false, af, false, bfl, (short)0, accl[t], false, false);
            v16h bfr = fr[(kk * 8 + t) * 32 + lane];
            accr[t] = __builtin_amdgcn_wmma_f32_16x16x32_f16(
                          false, af, false, bfr, (short)0, accr[t], false, false);
        }
    }

    // ---- store D + bias ----
    #pragma unroll
    for (int t = 0; t < 8; ++t) {
        int n  = t * 16 + col16;
        float bnl = bl[n];
        float bnr = br[n];
        #pragma unroll
        for (int v = 0; v < 8; ++v) {
            int m = tm * 16 + v + hiHalf * 8;
            Yl[(size_t)m * HID + n] = accl[t][v] + bnl;
            Yr[(size_t)m * HID + n] = accr[t][v] + bnr;
        }
    }
}

// ---------------------------------------------------------------------------
// Per-layer init: accum = 0 (N*HID), denom = 0, lmax = -inf (N*H)
// ---------------------------------------------------------------------------
__global__ void layer_init(float* __restrict__ accum, float* __restrict__ denom,
                           float* __restrict__ lmax, int n)
{
    int idx = blockIdx.x * blockDim.x + threadIdx.x;
    if (idx < n * HID) accum[idx] = 0.0f;
    if (idx < n * H) { denom[idx] = 0.0f; lmax[idx] = -__builtin_inff(); }
}

// ---------------------------------------------------------------------------
// Edge pass 1: per-edge attention logits + segment max into lmax.
// One wave per edge; lane owns 4 channels (float4 gathers, L2-resident).
// ---------------------------------------------------------------------------
__global__ void __launch_bounds__(256)
edge_logits(const int* __restrict__ src, const int* __restrict__ dst,
            const float* __restrict__ edge_attr, const float* __restrict__ We,
            const float* __restrict__ att,
            const float* __restrict__ xl, const float* __restrict__ xr,
            float* __restrict__ logits, float* __restrict__ lmax, int nE)
{
    __shared__ float sWe[FE * HID];   // 8 KB
    __shared__ float sAtt[HID];       // flattened (H,C)
    for (int i = threadIdx.x; i < FE * HID; i += blockDim.x) sWe[i] = We[i];
    for (int i = threadIdx.x; i < HID; i += blockDim.x)      sAtt[i] = att[i];
    __syncthreads();

    int lane = threadIdx.x & 31;
    int e = blockIdx.x * (blockDim.x >> 5) + (threadIdx.x >> 5);
    if (e >= nE) return;

    int s = src[e], d = dst[e];
    int c0 = lane * 4;

    float4 a4 = *(const float4*)(xl + (size_t)s * HID + c0);
    float4 b4 = *(const float4*)(xr + (size_t)d * HID + c0);
    float m0 = a4.x + b4.x, m1 = a4.y + b4.y, m2 = a4.z + b4.z, m3 = a4.w + b4.w;

    const float* ea = edge_attr + (size_t)e * FE;
    #pragma unroll
    for (int k = 0; k < FE; ++k) {
        float av = ea[k];                          // uniform per wave (broadcast)
        const float* wrow = &sWe[k * HID + c0];
        m0 = fmaf(av, wrow[0], m0);
        m1 = fmaf(av, wrow[1], m1);
        m2 = fmaf(av, wrow[2], m2);
        m3 = fmaf(av, wrow[3], m3);
    }
    // leaky relu then dot with att
    m0 = (m0 > 0.0f) ? m0 : NEG_SLOPE * m0;
    m1 = (m1 > 0.0f) ? m1 : NEG_SLOPE * m1;
    m2 = (m2 > 0.0f) ? m2 : NEG_SLOPE * m2;
    m3 = (m3 > 0.0f) ? m3 : NEG_SLOPE * m3;
    float part = m0 * sAtt[c0] + m1 * sAtt[c0 + 1] + m2 * sAtt[c0 + 2] + m3 * sAtt[c0 + 3];

    // reduce across 8 lanes of each head (lanes [8h, 8h+8))
    part += __shfl_xor(part, 1);
    part += __shfl_xor(part, 2);
    part += __shfl_xor(part, 4);

    if ((lane & 7) == 0) {
        int h = lane >> 3;
        logits[(size_t)e * H + h] = part;
        atomicMaxFloat(&lmax[(size_t)d * H + h], part);
    }
}

// ---------------------------------------------------------------------------
// Edge pass 2: ex = exp(logit - lmax[dst]); denom[dst,h] += ex;
//              accum[dst,:] += ex * xl[src,:]   (unnormalized scatter)
// ---------------------------------------------------------------------------
__global__ void __launch_bounds__(256)
edge_scatter(const int* __restrict__ src, const int* __restrict__ dst,
             const float* __restrict__ xl, const float* __restrict__ logits,
             const float* __restrict__ lmax,
             float* __restrict__ denom, float* __restrict__ accum, int nE)
{
    int lane = threadIdx.x & 31;
    int e = blockIdx.x * (blockDim.x >> 5) + (threadIdx.x >> 5);
    if (e >= nE) return;

    int s = src[e], d = dst[e];
    int h = lane >> 3;
    int c0 = lane * 4;

    float ex = __expf(logits[(size_t)e * H + h] - lmax[(size_t)d * H + h]);
    if ((lane & 7) == 0) atomicAdd(&denom[(size_t)d * H + h], ex);

    float4 a4 = *(const float4*)(xl + (size_t)s * HID + c0);
    float* outp = accum + (size_t)d * HID + c0;
    atomicAdd(outp + 0, ex * a4.x);
    atomicAdd(outp + 1, ex * a4.y);
    atomicAdd(outp + 2, ex * a4.z);
    atomicAdd(outp + 3, ex * a4.w);
}

// ---------------------------------------------------------------------------
// Node finalize: out = ELU(accum / (denom + 1e-16) + bias)   (in place OK)
// ---------------------------------------------------------------------------
__global__ void node_finalize(const float* __restrict__ accum,
                              const float* __restrict__ denom,
                              const float* __restrict__ bias,
                              float* __restrict__ hout, int n)
{
    int idx = blockIdx.x * blockDim.x + threadIdx.x;
    if (idx >= n * HID) return;
    int node = idx / HID;
    int c    = idx - node * HID;
    int h    = c >> 5;                            // C == 32
    float v = accum[idx] / (denom[node * H + h] + 1e-16f) + bias[c];
    v = (v > 0.0f) ? v : (__expf(v) - 1.0f);      // ELU
    hout[idx] = v;
}

// ---------------------------------------------------------------------------
// Pooling: per-graph mean + max over nodes (batch[] is sorted).
// blockDim = 128 (one channel/thread); block scans a node range and flushes
// per-graph partials with one atomic per (block, graph-boundary).
// ---------------------------------------------------------------------------
__global__ void pool_init(float* __restrict__ sums, float* __restrict__ maxs,
                          float* __restrict__ cnt)
{
    int idx = blockIdx.x * blockDim.x + threadIdx.x;
    if (idx < NB * HID) { sums[idx] = 0.0f; maxs[idx] = -__builtin_inff(); }
    if (idx < NB) cnt[idx] = 0.0f;
}

__global__ void __launch_bounds__(128)
pool_reduce(const float* __restrict__ hfeat, const int* __restrict__ batch,
            float* __restrict__ sums, float* __restrict__ maxs,
            float* __restrict__ cnt, int n)
{
    const int NPB = 512;
    int c  = threadIdx.x;
    int n0 = blockIdx.x * NPB;
    int n1 = min(n0 + NPB, n);

    int curB = -1;
    float sAcc = 0.0f, mAcc = -__builtin_inff();
    int cCnt = 0;

    for (int node = n0; node < n1; ++node) {
        int b = batch[node];                       // uniform broadcast
        if (b != curB) {
            if (curB >= 0) {
                atomicAdd(&sums[curB * HID + c], sAcc);
                atomicMaxFloat(&maxs[curB * HID + c], mAcc);
                if (c == 0) atomicAdd(&cnt[curB], (float)cCnt);
            }
            curB = b; sAcc = 0.0f; mAcc = -__builtin_inff(); cCnt = 0;
        }
        float v = hfeat[(size_t)node * HID + c];
        sAcc += v;
        mAcc = fmaxf(mAcc, v);
        ++cCnt;
    }
    if (curB >= 0) {
        atomicAdd(&sums[curB * HID + c], sAcc);
        atomicMaxFloat(&maxs[curB * HID + c], mAcc);
        if (c == 0) atomicAdd(&cnt[curB], (float)cCnt);
    }
}

__global__ void pool_emb(const float* __restrict__ sums, const float* __restrict__ maxs,
                         const float* __restrict__ cnt, float* __restrict__ emb)
{
    int idx = blockIdx.x * blockDim.x + threadIdx.x;   // NB * 2*HID
    if (idx >= NB * 2 * HID) return;
    int b = idx / (2 * HID);
    int c = idx - b * 2 * HID;
    float v;
    if (c < HID) {
        float ct = fmaxf(cnt[b], 1.0f);
        v = sums[b * HID + c] / ct;
    } else {
        float m = maxs[b * HID + (c - HID)];
        v = (m > -3.0e38f) ? m : 0.0f;                  // isfinite -> 0 guard
    }
    emb[idx] = v;
}

// ---------------------------------------------------------------------------
// Final MLP head: [B,512] @ Wf1(512x128) -> ReLU -> @ Wf2(128x1) -> sigmoid
// One block of 128 threads; B=8.
// ---------------------------------------------------------------------------
__global__ void __launch_bounds__(128)
head_mlp(const float* __restrict__ emb_a, const float* __restrict__ emb_b,
         const float* __restrict__ Wf1, const float* __restrict__ bf1,
         const float* __restrict__ Wf2, const float* __restrict__ bf2,
         float* __restrict__ out)
{
    __shared__ float comb[NB][4 * HID];   // 16 KB
    __shared__ float red[128];
    int t = threadIdx.x;

    for (int b = 0; b < NB; ++b) {
        for (int i = t; i < 2 * HID; i += 128) {
            comb[b][i]            = emb_a[b * 2 * HID + i];
            comb[b][2 * HID + i]  = emb_b[b * 2 * HID + i];
        }
    }
    __syncthreads();

    for (int b = 0; b < NB; ++b) {
        float acc = bf1[t];
        for (int k = 0; k < 4 * HID; ++k)
            acc = fmaf(comb[b][k], Wf1[k * HID + t], acc);
        acc = fmaxf(acc, 0.0f);                  // ReLU
        red[t] = acc * Wf2[t];
        __syncthreads();
        for (int s = 64; s > 0; s >>= 1) {
            if (t < s) red[t] += red[t + s];
            __syncthreads();
        }
        if (t == 0) out[b] = 1.0f / (1.0f + __expf(-(red[0] + bf2[0])));
        __syncthreads();
    }
}

// ---------------------------------------------------------------------------
// Host orchestration
// ---------------------------------------------------------------------------
extern "C" void kernel_launch(void* const* d_in, const int* in_sizes, int n_in,
                              void* d_out, int out_size, void* d_ws, size_t ws_size,
                              hipStream_t stream)
{
    (void)in_sizes; (void)n_in; (void)out_size; (void)ws_size;

    const float* x_a     = (const float*)d_in[0];
    const int*   ei_a    = (const int*)  d_in[1];
    const float* ea_a    = (const float*)d_in[2];
    const int*   batch_a = (const int*)  d_in[3];
    const float* x_b     = (const float*)d_in[4];
    const int*   ei_b    = (const int*)  d_in[5];
    const float* ea_b    = (const float*)d_in[6];
    const int*   batch_b = (const int*)  d_in[7];

    const float* W1l = (const float*)d_in[8];  const float* b1l = (const float*)d_in[9];
    const float* W1r = (const float*)d_in[10]; const float* b1r = (const float*)d_in[11];
    const float* W1e = (const float*)d_in[12]; const float* att1 = (const float*)d_in[13];
    const float* bias1 = (const float*)d_in[14];
    const float* W2l = (const float*)d_in[15]; const float* b2l = (const float*)d_in[16];
    const float* W2r = (const float*)d_in[17]; const float* b2r = (const float*)d_in[18];
    const float* W2e = (const float*)d_in[19]; const float* att2 = (const float*)d_in[20];
    const float* bias2 = (const float*)d_in[21];
    const float* Wf1 = (const float*)d_in[22]; const float* bf1 = (const float*)d_in[23];
    const float* Wf2 = (const float*)d_in[24]; const float* bf2 = (const float*)d_in[25];
    float* out = (float*)d_out;

    // workspace layout
    char* ws = (char*)d_ws;
    size_t off = 0;
    auto alloc = [&](size_t bytes) -> void* {
        void* p = (void*)(ws + off);
        off += (bytes + 255) & ~(size_t)255;
        return p;
    };
    float* xl     = (float*)alloc((size_t)N_NODES * HID * 4);
    float* xr     = (float*)alloc((size_t)N_NODES * HID * 4);
    float* buf0   = (float*)alloc((size_t)N_NODES * HID * 4);
    float* buf1   = (float*)alloc((size_t)N_NODES * HID * 4);
    float* logits = (float*)alloc((size_t)N_EDGES * H * 4);
    float* lmax   = (float*)alloc((size_t)N_NODES * H * 4);
    float* denom  = (float*)alloc((size_t)N_NODES * H * 4);
    float* sums   = (float*)alloc((size_t)NB * HID * 4);
    float* maxs   = (float*)alloc((size_t)NB * HID * 4);
    float* cnt    = (float*)alloc((size_t)NB * 4);
    float* emb_a  = (float*)alloc((size_t)NB * 2 * HID * 4);
    float* emb_b  = (float*)alloc((size_t)NB * 2 * HID * 4);
    _Float16* Wp1l = (_Float16*)alloc((size_t)HID * HID * 2);   // swizzled f16 weights
    _Float16* Wp1r = (_Float16*)alloc((size_t)HID * HID * 2);
    _Float16* Wp2l = (_Float16*)alloc((size_t)HID * HID * 2);
    _Float16* Wp2r = (_Float16*)alloc((size_t)HID * HID * 2);

    const int gemmBlocks = (N_NODES / 16 + 3) / 4;              // 4 waves/block
    const int initBlocks = (N_NODES * HID + 255) / 256;
    const int edgeBlocks = (N_EDGES + 7) / 8;                   // 8 waves/block
    const int poolBlocks = (N_NODES + 511) / 512;

    // Pre-swizzle all four node-transform weights into WMMA B-fragment layout.
    weight_swizzle<<<4, 256, 0, stream>>>(W1l, Wp1l);
    weight_swizzle<<<4, 256, 0, stream>>>(W1r, Wp1r);
    weight_swizzle<<<4, 256, 0, stream>>>(W2l, Wp2l);
    weight_swizzle<<<4, 256, 0, stream>>>(W2r, Wp2r);

    auto run_layer = [&](const float* Xin, const int* src, const int* dst,
                         const float* eattr,
                         const _Float16* Wpl, const float* bl,
                         const _Float16* Wpr, const float* br,
                         const float* We, const float* att, const float* bias,
                         float* accum) {
        gemm_node_wmma_dual<<<gemmBlocks, 128, 0, stream>>>(Xin, Wpl, Wpr, bl, br,
                                                            xl, xr, N_NODES);
        layer_init<<<initBlocks, 256, 0, stream>>>(accum, denom, lmax, N_NODES);
        edge_logits<<<edgeBlocks, 256, 0, stream>>>(src, dst, eattr, We, att,
                                                    xl, xr, logits, lmax, N_EDGES);
        edge_scatter<<<edgeBlocks, 256, 0, stream>>>(src, dst, xl, logits, lmax,
                                                     denom, accum, N_EDGES);
        node_finalize<<<initBlocks, 256, 0, stream>>>(accum, denom, bias, accum, N_NODES);
    };

    auto run_graph = [&](const float* X0, const int* ei, const float* eattr,
                         const int* batch, float* emb) {
        const int* src = ei;
        const int* dst = ei + N_EDGES;
        run_layer(X0,   src, dst, eattr, Wp1l, b1l, Wp1r, b1r, W1e, att1, bias1, buf0);
        run_layer(buf0, src, dst, eattr, Wp2l, b2l, Wp2r, b2r, W2e, att2, bias2, buf1);
        pool_init<<<(NB * HID + 255) / 256, 256, 0, stream>>>(sums, maxs, cnt);
        pool_reduce<<<poolBlocks, 128, 0, stream>>>(buf1, batch, sums, maxs, cnt, N_NODES);
        pool_emb<<<(NB * 2 * HID + 255) / 256, 256, 0, stream>>>(sums, maxs, cnt, emb);
    };

    run_graph(x_a, ei_a, ea_a, batch_a, emb_a);
    run_graph(x_b, ei_b, ea_b, batch_b, emb_b);
    head_mlp<<<1, 128, 0, stream>>>(emb_a, emb_b, Wf1, bf1, Wf2, bf2, out);
}
